// EntropyComponent_27530740367433
// MI455X (gfx1250) — matrix operand order repaired and checked by reading, not compiled, not run on hardware
//
#include <hip/hip_runtime.h>
#include <hip/hip_bf16.h>
#include <math.h>

#define DEV __device__ __forceinline__

typedef __attribute__((ext_vector_type(16))) _Float16 v16h;
typedef __attribute__((ext_vector_type(8)))  float    v8f;

// ---- model dims (match reference) ----
#define BB     4
#define LL     4096
#define DRAW   1024
#define HID    256
#define DINNER 512
#define DSTATE 64
#define PDIM   64
#define NHEADS 8
#define SS     1024          // L/4 after downsample
#define DXBC   640           // DINNER + 2*DSTATE
#define DPROJ  1160          // 2*DINNER + 2*DSTATE + NHEADS
#define HD2    128           // attention head dim (HID/2)

DEV float gelu_f(float x) {            // tanh-approx gelu (jax default)
    const float c = 0.7978845608028654f;
    float x3 = x * x * x;
    float t = tanhf(c * (x + 0.044715f * x3));
    return 0.5f * x * (1.f + t);
}
DEV float silu_f(float x) { return x / (1.f + __expf(-x)); }
DEV float softplus_f(float x) { return (x > 20.f) ? x : log1pf(__expf(x)); }

enum { EPI_NONE = 0, EPI_BIAS = 1, EPI_BIAS_GELU = 2, EPI_BIAS_RES = 3, EPI_RES = 4 };

// ---------------------------------------------------------------------------
// WMMA GEMM: C = epi(A[MxK](lda) * W + bias, res)
//   W is KxN (ldb) normally, NxK (ldb) when transB=1.
// block = 128 threads (4 waves), tile 128x64, BK = 32.
// Each wave computes a 32x64 strip: 2 A fragments x 4 B fragments -> 8 WMMAs
// per K-step (each B fragment feeds 2 WMMAs). Staging uses float4 global
// loads (global_load_b128) with packed f16x2 LDS stores.
// Requires: M % 128 == 0, K % 32 == 0, N % 4 == 0 (guards at float4 level).
// ---------------------------------------------------------------------------
__global__ __launch_bounds__(128)
void gemm_f16wmma_kernel(const float* __restrict__ A, int lda,
                         const float* __restrict__ Wm, int ldb, int transB,
                         float* __restrict__ C, int ldc,
                         int M, int N, int K,
                         const float* __restrict__ bias,
                         const float* __restrict__ res, int epi)
{
    __shared__ _Float16 As[128][40];   // [row][k], padded
    __shared__ _Float16 Bs[64][40];    // [col][k], padded (k-contiguous)

    const int tid  = threadIdx.x;
    const int lane = tid & 31;
    const int wave = tid >> 5;
    const int n0 = blockIdx.x * 64;
    const int m0 = blockIdx.y * 128;

    union CF { v8f v; float f[8]; } acc[2][4];
#pragma unroll
    for (int mi = 0; mi < 2; ++mi)
#pragma unroll
        for (int nt = 0; nt < 4; ++nt)
#pragma unroll
            for (int j = 0; j < 8; ++j) acc[mi][nt].f[j] = 0.f;

    // staging coordinates
    const int arow = tid >> 3;            // 0..15 (8 f4-lanes per row)
    const int acol = (tid & 7) * 4;       // 0..28
    const int kt_num = K >> 5;

    for (int kt = 0; kt < kt_num; ++kt) {
        const int k0 = kt << 5;

        // ---- stage A tile 128x32: 8 x float4 per thread ----
#pragma unroll
        for (int i = 0; i < 8; ++i) {
            int r = arow + i * 16;
            float4 va = *reinterpret_cast<const float4*>(
                &A[(size_t)(m0 + r) * lda + (k0 + acol)]);
            union { _Float16 h[4]; unsigned u[2]; } pk;
            pk.h[0] = (_Float16)va.x; pk.h[1] = (_Float16)va.y;
            pk.h[2] = (_Float16)va.z; pk.h[3] = (_Float16)va.w;
            *reinterpret_cast<unsigned*>(&As[r][acol])     = pk.u[0];
            *reinterpret_cast<unsigned*>(&As[r][acol + 2]) = pk.u[1];
        }

        // ---- stage B tile 32x64 -> Bs[col][k]: 4 x float4 per thread ----
        if (!transB) {
#pragma unroll
            for (int i = 0; i < 4; ++i) {
                int k = (tid >> 4) + i * 8;       // 0..31
                int n = (tid & 15) * 4;           // 0..60
                float4 vb = make_float4(0.f, 0.f, 0.f, 0.f);
                if (n0 + n < N)
                    vb = *reinterpret_cast<const float4*>(
                        &Wm[(size_t)(k0 + k) * ldb + (n0 + n)]);
                Bs[n + 0][k] = (_Float16)vb.x;
                Bs[n + 1][k] = (_Float16)vb.y;
                Bs[n + 2][k] = (_Float16)vb.z;
                Bs[n + 3][k] = (_Float16)vb.w;
            }
        } else {
#pragma unroll
            for (int i = 0; i < 4; ++i) {
                int n = (tid >> 3) + i * 16;      // 0..63
                int k = (tid & 7) * 4;            // 0..28
                float4 vb = make_float4(0.f, 0.f, 0.f, 0.f);
                if (n0 + n < N)
                    vb = *reinterpret_cast<const float4*>(
                        &Wm[(size_t)(n0 + n) * ldb + (k0 + k)]);
                union { _Float16 h[4]; unsigned u[2]; } pk;
                pk.h[0] = (_Float16)vb.x; pk.h[1] = (_Float16)vb.y;
                pk.h[2] = (_Float16)vb.z; pk.h[3] = (_Float16)vb.w;
                *reinterpret_cast<unsigned*>(&Bs[n][k])     = pk.u[0];
                *reinterpret_cast<unsigned*>(&Bs[n][k + 2]) = pk.u[1];
            }
        }
        if (kt + 1 < kt_num)   // cacheline hint for next A tile (global_prefetch)
            __builtin_prefetch(&A[(size_t)(m0 + arow) * lda + (k0 + 32 + acol)], 0, 1);
        __syncthreads();

        // ---- compute: wave w owns rows [w*32, w*32+32) ----
        // A fragment 16x32 f16 ISA layout: lane 0-15: M=lane, V0-3 K=0..7 pairs,
        // V4-7 K=16..23; lanes 16-31 shift K by +8.
        union AF { v16h v; _Float16 h[16]; } af0, af1, bf;
        const int khalf = lane >> 4;
        const int mrow0 = wave * 32 + (lane & 15);
        const int mrow1 = mrow0 + 16;
#pragma unroll
        for (int vv = 0; vv < 8; ++vv) {
            int kb = (vv < 4 ? 2 * vv : 16 + 2 * (vv - 4)) + 8 * khalf;
            af0.h[2 * vv]     = As[mrow0][kb];
            af0.h[2 * vv + 1] = As[mrow0][kb + 1];
            af1.h[2 * vv]     = As[mrow1][kb];
            af1.h[2 * vv + 1] = As[mrow1][kb + 1];
        }
#pragma unroll
        for (int nt = 0; nt < 4; ++nt) {
            const int ncol = nt * 16 + (lane & 15);
#pragma unroll
            for (int vv = 0; vv < 8; ++vv) {
                int kb = (vv < 4 ? 2 * vv : 16 + 2 * (vv - 4)) + 8 * khalf;  // B: same K order as A (lanes 16-31 shift K by +8)
                bf.h[2 * vv]     = Bs[ncol][kb];
                bf.h[2 * vv + 1] = Bs[ncol][kb + 1];
            }
            acc[0][nt].v = __builtin_amdgcn_wmma_f32_16x16x32_f16(
                false, af0.v, false, bf.v, (short)0, acc[0][nt].v, false, false);
            acc[1][nt].v = __builtin_amdgcn_wmma_f32_16x16x32_f16(
                false, af1.v, false, bf.v, (short)0, acc[1][nt].v, false, false);
        }
        __syncthreads();
    }

    // ---- epilogue: C/D layout: lane<16 -> N=lane, M=r; lane>=16 -> M=8+r ----
#pragma unroll
    for (int mi = 0; mi < 2; ++mi) {
#pragma unroll
        for (int nt = 0; nt < 4; ++nt) {
            int gc = n0 + nt * 16 + (lane & 15);
            if (gc >= N) continue;
            float bv = ((epi == EPI_BIAS || epi == EPI_BIAS_GELU || epi == EPI_BIAS_RES) && bias)
                           ? bias[gc] : 0.f;
#pragma unroll
            for (int r = 0; r < 8; ++r) {
                int gr = m0 + wave * 32 + mi * 16 + ((lane >> 4) << 3) + r;
                float v = acc[mi][nt].f[r] + bv;
                if (epi == EPI_BIAS_GELU) v = gelu_f(v);
                if ((epi == EPI_BIAS_RES || epi == EPI_RES) && res)
                    v += res[(size_t)gr * ldc + gc];
                C[(size_t)gr * ldc + gc] = v;
            }
        }
    }
}

// ---------------------------------------------------------------------------
// depthwise conv k=7 (same pad) fused with LayerNorm over 256 channels
// ---------------------------------------------------------------------------
__global__ __launch_bounds__(256)
void dwconv7_ln_kernel(const float* __restrict__ h, const float* __restrict__ w,
                       const float* __restrict__ g, const float* __restrict__ beta,
                       float* __restrict__ out)
{
    __shared__ float sbuf[256];
    const int c = threadIdx.x;
    const int pos = blockIdx.x;
    const int b = pos / LL, l = pos % LL;
    const size_t base = (size_t)b * LL * HID;
    float acc = 0.f;
#pragma unroll
    for (int j = 0; j < 7; ++j) {
        int ll = l + j - 3;
        if (ll >= 0 && ll < LL) acc += h[base + (size_t)ll * HID + c] * w[j * HID + c];
    }
    sbuf[c] = acc; __syncthreads();
    for (int s = 128; s > 0; s >>= 1) { if (c < s) sbuf[c] += sbuf[c + s]; __syncthreads(); }
    float mean = sbuf[0] * (1.f / HID); __syncthreads();
    float d = acc - mean;
    sbuf[c] = d * d; __syncthreads();
    for (int s = 128; s > 0; s >>= 1) { if (c < s) sbuf[c] += sbuf[c + s]; __syncthreads(); }
    float var = sbuf[0] * (1.f / HID);
    out[(size_t)pos * HID + c] = d * rsqrtf(var + 1e-5f) * g[c] + beta[c];
}

__global__ __launch_bounds__(256)
void ln256_kernel(const float* __restrict__ x, const float* __restrict__ g,
                  const float* __restrict__ beta, float* __restrict__ out)
{
    __shared__ float sbuf[256];
    int c = threadIdx.x; size_t row = blockIdx.x;
    float v = x[row * HID + c];
    sbuf[c] = v; __syncthreads();
    for (int s = 128; s > 0; s >>= 1) { if (c < s) sbuf[c] += sbuf[c + s]; __syncthreads(); }
    float mean = sbuf[0] * (1.f / HID); __syncthreads();
    float d = v - mean;
    sbuf[c] = d * d; __syncthreads();
    for (int s = 128; s > 0; s >>= 1) { if (c < s) sbuf[c] += sbuf[c + s]; __syncthreads(); }
    float var = sbuf[0] * (1.f / HID);
    out[row * HID + c] = d * rsqrtf(var + 1e-5f) * g[c] + beta[c];
}

__global__ __launch_bounds__(256)
void rms256_kernel(const float* __restrict__ x, const float* __restrict__ w,
                   float* __restrict__ out)
{
    __shared__ float sbuf[256];
    int t = threadIdx.x; size_t row = blockIdx.x;
    float v = x[row * HID + t];
    sbuf[t] = v * v; __syncthreads();
    for (int s = 128; s > 0; s >>= 1) { if (t < s) sbuf[t] += sbuf[t + s]; __syncthreads(); }
    float sc = rsqrtf(sbuf[0] * (1.f / HID) + 1e-6f);
    out[row * HID + t] = v * sc * w[t];
}

__global__ void dt_softplus_kernel(const float* __restrict__ zx,
                                   const float* __restrict__ dtb,
                                   float* __restrict__ dt)
{
    int idx = blockIdx.x * blockDim.x + threadIdx.x;
    if (idx >= BB * SS * NHEADS) return;
    int hh = idx & 7; size_t row = (size_t)(idx >> 3);
    dt[idx] = softplus_f(zx[row * DPROJ + (DPROJ - NHEADS) + hh] + dtb[hh]);
}

// causal depthwise conv k=4 (pad left 3) + bias + silu over 640 channels
__global__ __launch_bounds__(256)
void xbc_conv_kernel(const float* __restrict__ zx, const float* __restrict__ w,
                     const float* __restrict__ bias, float* __restrict__ out)
{
    size_t row = blockIdx.x;
    int b = (int)(row / SS), s = (int)(row % SS);
    for (int c = threadIdx.x; c < DXBC; c += 256) {
        float acc = bias[c];
#pragma unroll
        for (int j = 0; j < 4; ++j) {
            int ss2 = s - 3 + j;
            if (ss2 >= 0)
                acc += zx[((size_t)b * SS + ss2) * DPROJ + DINNER + c] * w[j * DXBC + c];
        }
        out[row * DXBC + c] = silu_f(acc);
    }
}

// ---------------------------------------------------------------------------
// selective scan: block = (b, head). 256 threads = 64 p x 4 n-groups,
// 16 state registers per thread; reduce over n-groups via __shfl_xor (wave32)
// ---------------------------------------------------------------------------
__global__ __launch_bounds__(256)
void ssm_scan_kernel(const float* __restrict__ xbc, const float* __restrict__ dt,
                     const float* __restrict__ A_log, const float* __restrict__ Dp,
                     float* __restrict__ ys)
{
    const int bh = blockIdx.x;
    const int b = bh >> 3, hh = bh & 7;
    const int t = threadIdx.x;
    const int p = t >> 2, gg = t & 3;
    __shared__ float sx[64], sB[64], sC[64], sdt[1];
    const float Ac = -__expf(A_log[hh]);
    const float Dc = Dp[hh];
    float st[16];
#pragma unroll
    for (int i = 0; i < 16; ++i) st[i] = 0.f;

    for (int s = 0; s < SS; ++s) {
        const size_t rb = ((size_t)b * SS + s) * DXBC;
        if (t < 64)            sx[t]        = xbc[rb + hh * 64 + t];
        else if (t < 128)      sB[t - 64]   = xbc[rb + 512 + (t - 64)];
        else if (t < 192)      sC[t - 128]  = xbc[rb + 576 + (t - 128)];
        else if (t == 192)     sdt[0]       = dt[((size_t)b * SS + s) * NHEADS + hh];
        __syncthreads();
        float dts = sdt[0];
        float da  = __expf(dts * Ac);
        float xb  = dts * sx[p];
        float part = 0.f;
#pragma unroll
        for (int i = 0; i < 16; ++i) {
            int n = gg * 16 + i;
            st[i] = st[i] * da + xb * sB[n];
            part += st[i] * sC[n];
        }
        part += __shfl_xor(part, 1);
        part += __shfl_xor(part, 2);
        if (gg == 0)
            ys[((size_t)b * SS + s) * DINNER + hh * 64 + p] = part + Dc * sx[p];
        __syncthreads();
    }
}

// y = rms(ys * silu(z), m_rms_w) over DINNER=512; z lives in zx rows (ld DPROJ)
__global__ __launch_bounds__(256)
void gate_rms_kernel(const float* __restrict__ ys, const float* __restrict__ zx,
                     const float* __restrict__ w, float* __restrict__ out)
{
    __shared__ float sbuf[256];
    size_t row = blockIdx.x; int t = threadIdx.x;
    float v0 = ys[row * DINNER + t]       * silu_f(zx[row * DPROJ + t]);
    float v1 = ys[row * DINNER + t + 256] * silu_f(zx[row * DPROJ + t + 256]);
    sbuf[t] = v0 * v0 + v1 * v1; __syncthreads();
    for (int s = 128; s > 0; s >>= 1) { if (t < s) sbuf[t] += sbuf[t + s]; __syncthreads(); }
    float sc = rsqrtf(sbuf[0] * (1.f / DINNER) + 1e-6f);
    out[row * DINNER + t]       = v0 * sc * w[t];
    out[row * DINNER + t + 256] = v1 * sc * w[t + 256];
}

// in-place softmax over rows of 1024 with pre-scale
__global__ __launch_bounds__(256)
void softmax1024_kernel(float* __restrict__ att, float scale)
{
    __shared__ float sbuf[256];
    size_t row = blockIdx.x;
    float* p = att + row * 1024;
    int t = threadIdx.x;
    float v[4];
    float mx = -3.4e38f;
#pragma unroll
    for (int i = 0; i < 4; ++i) { v[i] = p[t + i * 256] * scale; mx = fmaxf(mx, v[i]); }
    sbuf[t] = mx; __syncthreads();
    for (int s = 128; s > 0; s >>= 1) { if (t < s) sbuf[t] = fmaxf(sbuf[t], sbuf[t + s]); __syncthreads(); }
    mx = sbuf[0]; __syncthreads();
    float sum = 0.f;
#pragma unroll
    for (int i = 0; i < 4; ++i) { v[i] = __expf(v[i] - mx); sum += v[i]; }
    sbuf[t] = sum; __syncthreads();
    for (int s = 128; s > 0; s >>= 1) { if (t < s) sbuf[t] += sbuf[t + s]; __syncthreads(); }
    float inv = 1.f / sbuf[0];
#pragma unroll
    for (int i = 0; i < 4; ++i) p[t + i * 256] = v[i] * inv;
}

// ---------------------------------------------------------------------------
static inline void gemm(hipStream_t st, const float* A, int lda,
                        const float* Wm, int ldb, int transB,
                        float* C, int ldc, int M, int N, int K,
                        const float* bias, const float* res, int epi)
{
    dim3 grid((N + 63) / 64, M / 128, 1);
    gemm_f16wmma_kernel<<<grid, dim3(128, 1, 1), 0, st>>>(
        A, lda, Wm, ldb, transB, C, ldc, M, N, K, bias, res, epi);
}

extern "C" void kernel_launch(void* const* d_in, const int* in_sizes, int n_in,
                              void* d_out, int out_size, void* d_ws, size_t ws_size,
                              hipStream_t stream)
{
    const float* x        = (const float*)d_in[0];
    const float* w_in     = (const float*)d_in[1];
    const float* b_in     = (const float*)d_in[2];
    const float* cb_dw    = (const float*)d_in[3];
    const float* cb_ln_g  = (const float*)d_in[4];
    const float* cb_ln_b  = (const float*)d_in[5];
    const float* cb_w1    = (const float*)d_in[6];
    const float* cb_b1    = (const float*)d_in[7];
    const float* cb_w2    = (const float*)d_in[8];
    const float* cb_b2    = (const float*)d_in[9];
    const float* w_ds     = (const float*)d_in[10];
    const float* m_in_w   = (const float*)d_in[11];
    const float* m_in_b   = (const float*)d_in[12];
    const float* m_conv_w = (const float*)d_in[13];
    const float* m_conv_b = (const float*)d_in[14];
    const float* m_dt_b   = (const float*)d_in[15];
    const float* m_A_log  = (const float*)d_in[16];
    const float* m_D      = (const float*)d_in[17];
    const float* m_rms_w  = (const float*)d_in[18];
    const float* m_out_w  = (const float*)d_in[19];
    const float* norm_w   = (const float*)d_in[20];
    const float* w_qkv    = (const float*)d_in[21];
    const float* b_qkv    = (const float*)d_in[22];
    const float* w_o      = (const float*)d_in[23];
    const float* b_o      = (const float*)d_in[24];
    const float* ln1_g    = (const float*)d_in[25];
    const float* ln1_b    = (const float*)d_in[26];
    const float* ff1_w    = (const float*)d_in[27];
    const float* ff1_b    = (const float*)d_in[28];
    const float* ff2_w    = (const float*)d_in[29];
    const float* ff2_b    = (const float*)d_in[30];
    const float* ln2_g    = (const float*)d_in[31];
    const float* ln2_b    = (const float*)d_in[32];
    const float* oln_g    = (const float*)d_in[33];
    const float* oln_b    = (const float*)d_in[34];
    (void)in_sizes; (void)n_in; (void)out_size; (void)ws_size;

    char* ws = (char*)d_ws;
    const size_t MB = 1ull << 20;
    // phase-disjoint aliasing (~102 MB total):
    float* T1  = (float*)(ws + 0);        // conv phase: 67.2 MB
    float* ZX  = (float*)(ws + 0);        // mamba: 19 MB
    float* XBC = (float*)(ws + 20 * MB);  // 10.5 MB
    float* YSb = (float*)(ws + 31 * MB);  // 8.4 MB
    float* YG  = (float*)(ws + 40 * MB);  // 8.4 MB
    float* DTb = (float*)(ws + 49 * MB);  // 0.13 MB
    float* T2  = (float*)(ws + 50 * MB);  // 4.2 MB
    float* ATT = (float*)(ws + 0);        // attn: 33.6 MB
    float* QKV = (float*)(ws + 34 * MB);  // 12.6 MB
    float* AO  = (float*)(ws + 47 * MB);  // 4.2 MB
    float* FF  = (float*)(ws + 52 * MB);  // 8.4 MB
    float* TMPA= (float*)(ws + 61 * MB);  // 4.2 MB
    float* H   = (float*)(ws + 68 * MB);  // 16.8 MB (conv phase)
    float* H3  = (float*)(ws + 68 * MB);  // attn phase reuse
    float* U   = (float*)(ws + 85 * MB);  // 16.8 MB (conv phase)
    float* HS  = (float*)(ws + 85 * MB);  // mamba reuse: 4.2 MB
    float* H2  = (float*)(ws + 90 * MB);  // 4.2 MB
    float* out = (float*)d_out;

    const int MTOK = BB * LL;   // 16384
    const int MS   = BB * SS;   // 4096

    // 1) input projection
    gemm(stream, x, DRAW, w_in, HID, 0, H, HID, MTOK, HID, DRAW, b_in, nullptr, EPI_BIAS);

    // 2) conv blocks
    for (int i = 0; i < 2; ++i) {
        dwconv7_ln_kernel<<<MTOK, 256, 0, stream>>>(
            H, cb_dw + (size_t)i * 7 * HID, cb_ln_g + i * HID, cb_ln_b + i * HID, U);
        gemm(stream, U, HID, cb_w1 + (size_t)i * HID * 1024, 1024, 0,
             T1, 1024, MTOK, 1024, HID, cb_b1 + i * 1024, nullptr, EPI_BIAS_GELU);
        gemm(stream, T1, 1024, cb_w2 + (size_t)i * 1024 * HID, HID, 0,
             H, HID, MTOK, HID, 1024, cb_b2 + i * HID, H, EPI_BIAS_RES);
    }

    // 3) stride-4 downsample conv == GEMM (4096 x 1024) x (1024 x 256)
    gemm(stream, H, 1024, w_ds, HID, 0, HS, HID, MS, HID, 1024, nullptr, nullptr, EPI_NONE);

    // 4) mamba in-projection (N=1160, guarded tiles)
    gemm(stream, HS, HID, m_in_w, DPROJ, 0, ZX, DPROJ, MS, DPROJ, HID, m_in_b, nullptr, EPI_BIAS);

    // 5) dt = softplus(raw + bias)
    dt_softplus_kernel<<<(MS * NHEADS + 255) / 256, 256, 0, stream>>>(ZX, m_dt_b, DTb);

    // 6) causal dwconv4 + bias + silu on xBC
    xbc_conv_kernel<<<MS, 256, 0, stream>>>(ZX, m_conv_w, m_conv_b, XBC);

    // 7) selective scan
    ssm_scan_kernel<<<BB * NHEADS, 256, 0, stream>>>(XBC, DTb, m_A_log, m_D, YSb);

    // 8) gate + rms over DINNER
    gate_rms_kernel<<<MS, 256, 0, stream>>>(YSb, ZX, m_rms_w, YG);

    // 9) out projection + residual, 10) rms
    gemm(stream, YG, DINNER, m_out_w, HID, 0, T2, HID, MS, HID, DINNER, nullptr, HS, EPI_RES);
    rms256_kernel<<<MS, 256, 0, stream>>>(T2, norm_w, H2);

    // 11) qkv
    gemm(stream, H2, HID, w_qkv, 3 * HID, 0, QKV, 3 * HID, MS, 3 * HID, HID, b_qkv, nullptr, EPI_BIAS);

    // 12) scores = q . k^T per (b, head), 13) softmax, 14) att . v
    for (int b = 0; b < BB; ++b)
        for (int h = 0; h < 2; ++h) {
            const float* q = QKV + (size_t)b * SS * 768 + h * HD2;
            const float* k = q + HID;   // k block starts 256 cols later
            gemm(stream, q, 768, k, 768, 1,
                 ATT + (size_t)(b * 2 + h) * SS * SS, SS, SS, SS, HD2,
                 nullptr, nullptr, EPI_NONE);
        }
    softmax1024_kernel<<<BB * 2 * SS, 256, 0, stream>>>(ATT, 0.08838834764831845f);
    for (int b = 0; b < BB; ++b)
        for (int h = 0; h < 2; ++h) {
            const float* v = QKV + (size_t)b * SS * 768 + 512 + h * HD2;
            gemm(stream, ATT + (size_t)(b * 2 + h) * SS * SS, SS, v, 768, 0,
                 AO + (size_t)b * SS * HID + h * HD2, HID, SS, HD2, SS,
                 nullptr, nullptr, EPI_NONE);
        }

    // 15) output proj + residual + ln1
    gemm(stream, AO, HID, w_o, HID, 0, TMPA, HID, MS, HID, HID, b_o, H2, EPI_BIAS_RES);
    ln256_kernel<<<MS, 256, 0, stream>>>(TMPA, ln1_g, ln1_b, H3);

    // 16-17) FFN
    gemm(stream, H3, HID, ff1_w, 2 * HID, 0, FF, 2 * HID, MS, 2 * HID, HID, ff1_b, nullptr, EPI_BIAS_GELU);
    gemm(stream, FF, 2 * HID, ff2_w, HID, 0, TMPA, HID, MS, HID, 2 * HID, ff2_b, H3, EPI_BIAS_RES);

    // 18) final layernorms (second one in-place on d_out)
    ln256_kernel<<<MS, 256, 0, stream>>>(TMPA, ln2_g, ln2_b, out);
    ln256_kernel<<<MS, 256, 0, stream>>>(out, oln_g, oln_b, out);
}
